// SequentialTransductionUnitJagged_34849364639842
// MI455X (gfx1250) — matrix operand order repaired and checked by reading, not compiled, not run on hardware
//
#include <hip/hip_runtime.h>
#include <hip/hip_bf16.h>

#define BB 4
#define NN 2048
#define DD 256
#define HH 4
#define DA 64
#define DL 64
#define FU 1024   // H*(2*DL+2*DA)

typedef __attribute__((ext_vector_type(16))) __bf16 bf16x16;
typedef __attribute__((ext_vector_type(8)))  float  f32x8;

static __device__ __forceinline__ f32x8 wmma_bf16(bf16x16 a, bf16x16 b, f32x8 c) {
    // D = A(16x32 bf16) x B(32x16 bf16) + C(16x16 f32)
    return __builtin_amdgcn_wmma_f32_16x16x32_bf16(false, a, false, b, (short)0, c, false, false);
}

static __device__ __forceinline__ float silu_f(float v) {
    return v / (1.0f + __expf(-v));
}

static __device__ __forceinline__ float wave_red_sum(float v) {
#pragma unroll
    for (int off = 16; off > 0; off >>= 1) v += __shfl_xor(v, off, 32);
    return v;
}

// ---------------- weight conversion (transpose to K-contiguous bf16) ----------
__global__ void stu_convert_w(const float* __restrict__ Wu, const float* __restrict__ Wo,
                              __bf16* __restrict__ WuT, __bf16* __restrict__ WoT) {
    int idx = blockIdx.x * blockDim.x + threadIdx.x;
    if (idx < DD * FU) {
        int kk = idx >> 10, j = idx & (FU - 1);        // Wu[k][j]
        WuT[(long)j * DD + kk] = (__bf16)Wu[idx];      // WuT[j][k]
    } else {
        int t = idx - DD * FU;
        if (t < DD * DD) {
            int kk = t >> 8, j = t & 255;              // Wo[k][j]
            WoT[(long)j * DD + kk] = (__bf16)Wo[t];
        }
    }
}

// ---------------- input layernorm -> bf16 ------------------------------------
__global__ void stu_ln_in(const float* __restrict__ x, const float* __restrict__ g,
                          const float* __restrict__ bta, __bf16* __restrict__ nx) {
    int lane = threadIdx.x & 31;
    long row = (long)blockIdx.x * 8 + (threadIdx.x >> 5);   // < B*N
    const float* xr = x + row * DD;
    float vals[8], s = 0.f, s2 = 0.f;
#pragma unroll
    for (int i = 0; i < 8; ++i) {
        float v = xr[lane * 8 + i];
        vals[i] = v; s += v; s2 += v * v;
    }
    s = wave_red_sum(s); s2 = wave_red_sum(s2);
    float mu  = s * (1.0f / DD);
    float var = s2 * (1.0f / DD) - mu * mu;
    float inv = rsqrtf(var + 1e-5f);
#pragma unroll
    for (int i = 0; i < 8; ++i) {
        int c = lane * 8 + i;
        nx[row * DD + c] = (__bf16)((vals[i] - mu) * inv * g[c] + bta[c]);
    }
}

// ---------------- GEMM1: uvqk = silu(nx @ Wuvqk + b), 16x64 tile per wave ----
__global__ __launch_bounds__(32) void stu_gemm1(const __bf16* __restrict__ nx,
                                                const __bf16* __restrict__ WuT,
                                                const float* __restrict__ b_uvqk,
                                                float* __restrict__ u,
                                                __bf16* __restrict__ q,
                                                __bf16* __restrict__ k,
                                                __bf16* __restrict__ vT) {
    int lane = threadIdx.x, l16 = lane & 15, lh = lane >> 4;
    int i0 = blockIdx.x * 16;       // rows in [0, B*N)
    int j0 = blockIdx.y * 64;       // cols in [0, FU), 4 x 16-col tiles
    int rowA = i0 + l16;
    f32x8 acc[4] = {f32x8{}, f32x8{}, f32x8{}, f32x8{}};
    for (int kk = 0; kk < DD; kk += 32) {
        bf16x16 a;
#pragma unroll
        for (int r = 0; r < 8; ++r) {
            int ka = kk + (lh << 4) + (r << 1);
            const __bf16* pa = nx + (long)rowA * DD + ka;
            a[2 * r] = pa[0]; a[2 * r + 1] = pa[1];
        }
#pragma unroll
        for (int jt = 0; jt < 4; ++jt) {
            int colB = j0 + jt * 16 + l16;
            bf16x16 b;
#pragma unroll
            for (int r = 0; r < 8; ++r) {
                int kb = kk + (lh << 4) + (r << 1);
                const __bf16* pb = WuT + (long)colB * DD + kb;
                b[2 * r] = pb[0]; b[2 * r + 1] = pb[1];
            }
            acc[jt] = wmma_bf16(a, b, acc[jt]);
        }
    }
#pragma unroll
    for (int jt = 0; jt < 4; ++jt) {
#pragma unroll
        for (int j = 0; j < 8; ++j) {
            int grow = i0 + j + (lh << 3);
            int gcol = j0 + jt * 16 + l16;
            float val = silu_f(acc[jt][j] + b_uvqk[gcol]);
            int bidx = grow >> 11, n = grow & (NN - 1);
            int chunk = gcol >> 8, cc = gcol & 255;
            int hh = cc >> 6, d = cc & 63;
            if (chunk == 0)      u[(long)grow * 256 + cc] = val;
            else if (chunk == 1) vT[(((long)(bidx * HH + hh) * DL + d) * NN) + n] = (__bf16)val;
            else if (chunk == 2) q[(((long)(bidx * HH + hh) * NN + n) * DA) + d] = (__bf16)val;
            else                 k[(((long)(bidx * HH + hh) * NN + n) * DA) + d] = (__bf16)val;
        }
    }
}

// ---------------- attention: flash-style wave-per-16-query-tile --------------
__global__ __launch_bounds__(32) void stu_attn(const __bf16* __restrict__ q,
                                               const __bf16* __restrict__ k,
                                               const __bf16* __restrict__ vT,
                                               const float* __restrict__ w_pos,
                                               const float* __restrict__ w_ts,
                                               const long long* __restrict__ lengths,
                                               const long long* __restrict__ timestamps,
                                               float* __restrict__ av) {
    int lane = threadIdx.x, l16 = lane & 15, lh = lane >> 4;
    int n0 = blockIdx.x * 16;
    int h  = blockIdx.y;
    int b  = blockIdx.z;
    const __bf16* qh = q  + (long)(b * HH + h) * NN * DA;
    const __bf16* kh = k  + (long)(b * HH + h) * NN * DA;
    const __bf16* vh = vT + (long)(b * HH + h) * DL * NN;
    const long long* tsb = timestamps + (long)b * NN;
    long long lenb = lengths[b];

    __shared__ __bf16 sattn[16 * 32];   // 16 query rows x 32 keys, bf16

    // Preload q A-fragments for K = DA = 64 (two 16x32 frags), fixed per tile.
    bf16x16 aq[2];
#pragma unroll
    for (int t = 0; t < 2; ++t)
#pragma unroll
        for (int r = 0; r < 8; ++r) {
            int d = t * 32 + (lh << 4) + (r << 1);
            const __bf16* p = qh + (long)(n0 + l16) * DA + d;
            aq[t][2 * r] = p[0]; aq[t][2 * r + 1] = p[1];
        }
    // Per-lane query-row timestamps in f32 (ts < 1e7 is exact in f32; bucket
    // resolution is log-scale so f32 diff is sufficient).
    float ts_n[8];
#pragma unroll
    for (int j = 0; j < 8; ++j) ts_n[j] = (float)tsb[n0 + j + (lh << 3)];

    f32x8 acc[4] = {f32x8{}, f32x8{}, f32x8{}, f32x8{}};

    for (int m0 = 0; m0 <= n0 + 15; m0 += 32) {
#pragma unroll
        for (int half = 0; half < 2; ++half) {
            int ms = m0 + half * 16;
            int m  = ms + l16;
            int msafe = m < NN ? m : NN - 1;
            f32x8 cqk = {};
#pragma unroll
            for (int t = 0; t < 2; ++t) {
                bf16x16 bk;
#pragma unroll
                for (int r = 0; r < 8; ++r) {
                    int d = t * 32 + (lh << 4) + (r << 1);
                    const __bf16* p = kh + (long)msafe * DA + d;
                    bk[2 * r] = p[0]; bk[2 * r + 1] = p[1];
                }
                cqk = wmma_bf16(aq[t], bk, cqk);
            }
            float tsm = (float)tsb[msafe];
#pragma unroll
            for (int j = 0; j < 8; ++j) {
                int n = n0 + j + (lh << 3);
                float pb = w_pos[n - msafe + (NN - 1)];
                float ad = fabsf(ts_n[j] - tsm);
                ad = ad > 1.0f ? ad : 1.0f;
                int bucket = (int)(__logf(ad) * (1.0f / 0.301f));
                bucket = bucket < 0 ? 0 : (bucket > 128 ? 128 : bucket);
                float a = silu_f(cqk[j] + pb + w_ts[bucket]) * (1.0f / NN);
                bool ok = (n >= m) && ((long long)m < lenb);
                sattn[(j + (lh << 3)) * 32 + half * 16 + l16] = (__bf16)(ok ? a : 0.0f);
            }
        }
        __syncthreads();
        bf16x16 aA;
#pragma unroll
        for (int r = 0; r < 8; ++r) {
            int kk = (lh << 4) + (r << 1);
            const __bf16* p = &sattn[l16 * 32 + kk];
            aA[2 * r] = p[0]; aA[2 * r + 1] = p[1];
        }
#pragma unroll
        for (int jt = 0; jt < 4; ++jt) {
            bf16x16 bV;
#pragma unroll
            for (int r = 0; r < 8; ++r) {
                int kk = (lh << 4) + (r << 1);
                int mg = m0 + kk;
                int msf = mg < NN - 1 ? mg : NN - 2;
                const __bf16* p = vh + (long)(jt * 16 + l16) * NN + msf;
                bV[2 * r] = p[0]; bV[2 * r + 1] = p[1];
            }
            acc[jt] = wmma_bf16(aA, bV, acc[jt]);
        }
        __syncthreads();
    }
#pragma unroll
    for (int jt = 0; jt < 4; ++jt)
#pragma unroll
        for (int j = 0; j < 8; ++j) {
            int n = n0 + j + (lh << 3);
            av[((long)b * NN + n) * 256 + h * DL + jt * 16 + l16] = acc[jt][j];
        }
}

// ---------------- layernorm(av) * u -> bf16 ----------------------------------
__global__ void stu_ln_attn(const float* __restrict__ av, const float* __restrict__ u,
                            const float* __restrict__ g, const float* __restrict__ bta,
                            __bf16* __restrict__ hbuf) {
    int lane = threadIdx.x & 31;
    long row = (long)blockIdx.x * 8 + (threadIdx.x >> 5);
    const float* ar = av + row * 256;
    float vals[8], s = 0.f, s2 = 0.f;
#pragma unroll
    for (int i = 0; i < 8; ++i) {
        float v = ar[lane * 8 + i];
        vals[i] = v; s += v; s2 += v * v;
    }
    s = wave_red_sum(s); s2 = wave_red_sum(s2);
    float mu  = s * (1.0f / 256.0f);
    float var = s2 * (1.0f / 256.0f) - mu * mu;
    float inv = rsqrtf(var + 1e-5f);
#pragma unroll
    for (int i = 0; i < 8; ++i) {
        int c = lane * 8 + i;
        hbuf[row * 256 + c] = (__bf16)(((vals[i] - mu) * inv * g[c] + bta[c]) * u[row * 256 + c]);
    }
}

// ---------------- GEMM2: out = h @ Wo + b_o + x, 16x64 tile per wave ---------
__global__ __launch_bounds__(32) void stu_gemm2(const __bf16* __restrict__ hbuf,
                                                const __bf16* __restrict__ WoT,
                                                const float* __restrict__ b_o,
                                                const float* __restrict__ x,
                                                const long long* __restrict__ lengths,
                                                float* __restrict__ out) {
    int lane = threadIdx.x, l16 = lane & 15, lh = lane >> 4;
    int i0 = blockIdx.x * 16;
    int j0 = blockIdx.y * 64;
    int rowA = i0 + l16;
    f32x8 acc[4] = {f32x8{}, f32x8{}, f32x8{}, f32x8{}};
    for (int kk = 0; kk < 256; kk += 32) {
        bf16x16 a;
#pragma unroll
        for (int r = 0; r < 8; ++r) {
            int ka = kk + (lh << 4) + (r << 1);
            const __bf16* pa = hbuf + (long)rowA * 256 + ka;
            a[2 * r] = pa[0]; a[2 * r + 1] = pa[1];
        }
#pragma unroll
        for (int jt = 0; jt < 4; ++jt) {
            int colB = j0 + jt * 16 + l16;
            bf16x16 b;
#pragma unroll
            for (int r = 0; r < 8; ++r) {
                int kb = kk + (lh << 4) + (r << 1);
                const __bf16* pb = WoT + (long)colB * 256 + kb;
                b[2 * r] = pb[0]; b[2 * r + 1] = pb[1];
            }
            acc[jt] = wmma_bf16(a, b, acc[jt]);
        }
    }
#pragma unroll
    for (int jt = 0; jt < 4; ++jt) {
#pragma unroll
        for (int j = 0; j < 8; ++j) {
            int grow = i0 + j + (lh << 3);
            int gcol = j0 + jt * 16 + l16;
            int bidx = grow >> 11, n = grow & (NN - 1);
            float val = acc[jt][j] + b_o[gcol] + x[(long)grow * DD + gcol];
            bool ok = (long long)n < lengths[bidx];
            out[(long)grow * DD + gcol] = ok ? val : 0.0f;
        }
    }
}

extern "C" void kernel_launch(void* const* d_in, const int* in_sizes, int n_in,
                              void* d_out, int out_size, void* d_ws, size_t ws_size,
                              hipStream_t stream) {
    const float*     x       = (const float*)d_in[0];
    const float*     W_uvqk  = (const float*)d_in[1];
    const float*     b_uvqk  = (const float*)d_in[2];
    const float*     W_o     = (const float*)d_in[3];
    const float*     b_o     = (const float*)d_in[4];
    const float*     ln_in_g = (const float*)d_in[5];
    const float*     ln_in_b = (const float*)d_in[6];
    const float*     ln_at_g = (const float*)d_in[7];
    const float*     ln_at_b = (const float*)d_in[8];
    const float*     w_pos   = (const float*)d_in[9];
    const float*     w_ts    = (const float*)d_in[10];
    const long long* lengths = (const long long*)d_in[11];
    const long long* tstamps = (const long long*)d_in[12];
    float* out = (float*)d_out;

    char* w = (char*)d_ws;
    __bf16* WuT  = (__bf16*)w; w += (size_t)DD * FU * 2;     // 512 KB
    __bf16* WoT  = (__bf16*)w; w += (size_t)DD * DD * 2;     // 128 KB
    __bf16* nx   = (__bf16*)w; w += (size_t)BB * NN * DD * 2;// 4 MB
    float*  u    = (float*) w; w += (size_t)BB * NN * 256 * 4;   // 8 MB
    __bf16* qb   = (__bf16*)w; w += (size_t)BB * HH * NN * DA * 2; // 4 MB
    __bf16* kb   = (__bf16*)w; w += (size_t)BB * HH * NN * DA * 2; // 4 MB
    __bf16* vT   = (__bf16*)w; w += (size_t)BB * HH * DL * NN * 2; // 4 MB
    float*  av   = (float*) w; w += (size_t)BB * NN * 256 * 4;   // 8 MB
    __bf16* hbuf = (__bf16*)w; w += (size_t)BB * NN * 256 * 2;   // 4 MB

    // 1) weight conversion + transpose
    {
        int total = DD * FU + DD * DD;
        stu_convert_w<<<(total + 255) / 256, 256, 0, stream>>>(W_uvqk, W_o, WuT, WoT);
    }
    // 2) input layernorm
    stu_ln_in<<<(BB * NN) / 8, 256, 0, stream>>>(x, ln_in_g, ln_in_b, nx);
    // 3) GEMM1 (wave per 16x64 tile): M=8192, N=1024, K=256
    {
        dim3 grid(BB * NN / 16, FU / 64);
        stu_gemm1<<<grid, 32, 0, stream>>>(nx, WuT, b_uvqk, u, qb, kb, vT);
    }
    // 4) attention
    {
        dim3 grid(NN / 16, HH, BB);
        stu_attn<<<grid, 32, 0, stream>>>(qb, kb, vT, w_pos, w_ts, lengths, tstamps, av);
    }
    // 5) layernorm(av) * u
    stu_ln_attn<<<(BB * NN) / 8, 256, 0, stream>>>(av, u, ln_at_g, ln_at_b, hbuf);
    // 6) GEMM2 + residual + mask: M=8192, N=256, K=256
    {
        dim3 grid(BB * NN / 16, DD / 64);
        stu_gemm2<<<grid, 32, 0, stream>>>(hbuf, WoT, b_o, x, lengths, out);
    }
    (void)in_sizes; (void)n_in; (void)out_size; (void)ws_size;
}
